// ScaledDotAttention_29575144800949
// MI455X (gfx1250) — matrix operand order repaired and checked by reading, not compiled, not run on hardware
//
#include <hip/hip_runtime.h>
#include <math.h>

// ---------------- problem constants (fixed by the reference) ----------------
#define IN_C      64
#define OUT_C     128
#define HEADS     8
#define NORD      9
#define NENT      29
#define NPROJ     2
#define TILE_M    64      // 4 WMMA M-tiles per block; each B fragment feeds 4 tiles
#define NUM_NODES 10000   // reference setup: num_nodes = 10000

typedef __attribute__((ext_vector_type(16))) __bf16 v16bf;
typedef __attribute__((ext_vector_type(8)))  __bf16 v8bf;
typedef __attribute__((ext_vector_type(8)))  float  v8f;

// SO(2) path expansion: for each output order (0..8), the list of
// (input order, flat weight index, sign) contributions, grouped by o.
// Kept in __constant__ memory and indexed at runtime so the main loops stay
// rolled (full unroll blew past 256 VGPRs and spilled).
__constant__ int c_O_OFF[NORD + 1] = {0, 3, 7, 10, 14, 16, 20, 23, 27, 29};
__constant__ int c_ENT_I[NENT] = {
    0, 2, 6,              // o=0 (m=0)
    3, 7, 1, 5,           // o=1 (m=1, neg)
    0, 2, 6,              // o=2 (m=0)
    3, 7, 1, 5,           // o=3 (m=1, pos)
    8, 4,                 // o=4 (m=2, neg)
    3, 7, 1, 5,           // o=5 (m=1, neg)
    0, 2, 6,              // o=6 (m=0)
    3, 7, 1, 5,           // o=7 (m=1, pos)
    8, 4                  // o=8 (m=2, pos)
};
__constant__ int   g_ENT_W[NENT] = {
    0, 3, 6,
    10, 14, 9, 13,
    1, 4, 7,
    9, 13, 10, 14,
    18, 17,
    12, 16, 11, 15,
    2, 5, 8,
    11, 15, 12, 16,
    17, 18
};
__constant__ float g_ENT_S[NENT] = {
    1.f, 1.f, 1.f,
    1.f, 1.f, 1.f, 1.f,
    1.f, 1.f, 1.f,
    1.f, 1.f, -1.f, -1.f,
    1.f, 1.f,
    1.f, 1.f, 1.f, 1.f,
    1.f, 1.f, 1.f,
    1.f, 1.f, -1.f, -1.f,
    1.f, -1.f
};

// ---------------- helpers ----------------
__device__ inline unsigned short f32_bf16(float f) {
    unsigned int u = __float_as_uint(f);
    unsigned int r = (u + 0x7FFFu + ((u >> 16) & 1u)) >> 16; // RNE
    return (unsigned short)r;
}
__device__ inline unsigned int pack_bf16x2(float a, float b) {
    return (unsigned int)f32_bf16(a) | ((unsigned int)f32_bf16(b) << 16);
}

// A fragment (16x32 bf16): 16 contiguous elements per lane.
__device__ inline v16bf frag_a(const unsigned short* p) {
    return *(const v16bf*)(p);
}
// B fragment (32x16 bf16): 16 contiguous elements per lane.
__device__ inline v16bf frag_b(const unsigned short* p) {
    return *(const v16bf*)(p);
}

// XOR-butterfly sum over each 16-lane group via ds_swizzle
// (offset[15]=0: and=offset[4:0], or=offset[9:5], xor=offset[14:10]).
__device__ inline float red16(float v) {
    v += __int_as_float(__builtin_amdgcn_ds_swizzle(__float_as_int(v), 0x041F)); // xor 1
    v += __int_as_float(__builtin_amdgcn_ds_swizzle(__float_as_int(v), 0x081F)); // xor 2
    v += __int_as_float(__builtin_amdgcn_ds_swizzle(__float_as_int(v), 0x101F)); // xor 4
    v += __int_as_float(__builtin_amdgcn_ds_swizzle(__float_as_int(v), 0x201F)); // xor 8
    return v;
}

__device__ inline void atomicMaxF(float* addr, float val) {
    if (val >= 0.f) atomicMax((int*)addr, __float_as_int(val));
    else            atomicMin((unsigned int*)addr, (unsigned int)__float_as_int(val));
}

// ---------------- weight pack: f32 [w,c,col] -> signed bf16 [p,entry,col,c] --
__global__ __launch_bounds__(256) void so2_pack_w(const float* __restrict__ Wq,
                                                  const float* __restrict__ Wk,
                                                  unsigned short* __restrict__ wpack) {
    int i = blockIdx.x * blockDim.x + threadIdx.x;
    const int total = NPROJ * NENT * OUT_C * IN_C;
    if (i >= total) return;
    int c   = i & (IN_C - 1);
    int r   = i >> 6;
    int col = r & (OUT_C - 1);
    r >>= 7;
    int e = r % NENT;
    int p = r / NENT;
    const float* W = p ? Wk : Wq;
    float v = g_ENT_S[e] * W[(long)g_ENT_W[e] * (IN_C * OUT_C) + c * OUT_C + col];
    wpack[i] = f32_bf16(v);
}

// ---------------- main: SO(2) linear (q,k) + per-head dot, via WMMA ---------
// Block = 8 waves = 8 heads, 64 edges (four 16-row WMMA M-tiles per wave).
__global__ __launch_bounds__(256) void so2_attn_pre(const float* __restrict__ xq,
                                                    const float* __restrict__ xk,
                                                    const unsigned short* __restrict__ wpack,
                                                    float* __restrict__ pre, int E) {
    // [proj][M=64][ord][c] bf16 : 2*64*9*64*2 = 147,456 B (320 KB/WGP on CDNA5)
    __shared__ __align__(16) unsigned short ldsx[NPROJ * TILE_M * NORD * IN_C];

    const int  tid = threadIdx.x;
    const long e0  = (long)blockIdx.x * TILE_M;

    // Stage 64 edges of x_q / x_k into LDS as bf16 (flat coalesced copy,
    // zero-padded past E for the tail block).
    {
        const float4* xq4 = (const float4*)xq;
        const float4* xk4 = (const float4*)xk;
        const long base4 = e0 * (NORD * IN_C) / 4;
        const long lim4  = (long)E * (NORD * IN_C) / 4;
        uint2* dq = (uint2*)(ldsx);
        uint2* dk = (uint2*)(ldsx + TILE_M * NORD * IN_C);
        const int n4 = TILE_M * NORD * IN_C / 4; // 9216
        for (int i = tid; i < n4; i += 256) {
            long gi = base4 + i;
            float4 v = make_float4(0.f, 0.f, 0.f, 0.f);
            float4 w = make_float4(0.f, 0.f, 0.f, 0.f);
            if (gi < lim4) { v = xq4[gi]; w = xk4[gi]; }
            dq[i] = make_uint2(pack_bf16x2(v.x, v.y), pack_bf16x2(v.z, v.w));
            dk[i] = make_uint2(pack_bf16x2(w.x, w.y), pack_bf16x2(w.z, w.w));
        }
    }
    __syncthreads();

    const int lane = tid & 31;
    const int head = tid >> 5;          // wave w handles head w (one 16-col N tile)
    const int lrow = lane & 15;
    const int lhi  = lane >> 4;

    const unsigned short* ldsq = ldsx;
    const unsigned short* ldsk = ldsx + TILE_M * NORD * IN_C;
    const unsigned short* wq   = wpack;
    const unsigned short* wk   = wpack + (long)NENT * OUT_C * IN_C;
    const int tS = 16 * (NORD * IN_C);  // LDS stride between 16-edge M-tiles

    // per-lane invariant parts of the fragment addresses
    const int  abase = lrow * (NORD * IN_C) + 16 * lhi;
    const long bbase = (long)(head * 16 + lrow) * IN_C + 16 * lhi;

    v8f pre0 = {}, pre1 = {}, pre2 = {}, pre3 = {};
#pragma unroll 1
    for (int o = 0; o < NORD; ++o) {
        v8f qa0 = {}, qa1 = {}, qa2 = {}, qa3 = {};
        v8f ka0 = {}, ka1 = {}, ka2 = {}, ka3 = {};
        const int tend = c_O_OFF[o + 1];
#pragma unroll 1
        for (int t = c_O_OFF[o]; t < tend; ++t) {
            const int iord = c_ENT_I[t];
#pragma unroll
            for (int kh = 0; kh < 2; ++kh) {
                // A: lane holds M=lrow; elems -> c = kh*32+16*lhi+j (contiguous)
                const int  ab = abase + iord * IN_C + kh * 32;
                // B: lane holds N=head*16+lrow; elems -> c = kh*32+16*lhi+j (contiguous)
                const long wb = (long)t * (OUT_C * IN_C) + bbase + kh * 32;
                // one B fetch feeds four M-tiles (quarters L2 weight traffic)
                v16bf bq = frag_b(wq + wb);
                qa0 = __builtin_amdgcn_wmma_f32_16x16x32_bf16(false, frag_a(ldsq + ab), false, bq,
                                                              (short)0, qa0, false, false);
                qa1 = __builtin_amdgcn_wmma_f32_16x16x32_bf16(false, frag_a(ldsq + ab + tS), false, bq,
                                                              (short)0, qa1, false, false);
                qa2 = __builtin_amdgcn_wmma_f32_16x16x32_bf16(false, frag_a(ldsq + ab + 2 * tS), false, bq,
                                                              (short)0, qa2, false, false);
                qa3 = __builtin_amdgcn_wmma_f32_16x16x32_bf16(false, frag_a(ldsq + ab + 3 * tS), false, bq,
                                                              (short)0, qa3, false, false);
                v16bf bk = frag_b(wk + wb);
                ka0 = __builtin_amdgcn_wmma_f32_16x16x32_bf16(false, frag_a(ldsk + ab), false, bk,
                                                              (short)0, ka0, false, false);
                ka1 = __builtin_amdgcn_wmma_f32_16x16x32_bf16(false, frag_a(ldsk + ab + tS), false, bk,
                                                              (short)0, ka1, false, false);
                ka2 = __builtin_amdgcn_wmma_f32_16x16x32_bf16(false, frag_a(ldsk + ab + 2 * tS), false, bk,
                                                              (short)0, ka2, false, false);
                ka3 = __builtin_amdgcn_wmma_f32_16x16x32_bf16(false, frag_a(ldsk + ab + 3 * tS), false, bk,
                                                              (short)0, ka3, false, false);
            }
        }
#pragma unroll
        for (int r = 0; r < 8; ++r) {
            pre0[r] += qa0[r] * ka0[r];
            pre1[r] += qa1[r] * ka1[r];
            pre2[r] += qa2[r] * ka2[r];
            pre3[r] += qa3[r] * ka3[r];
        }
    }

    // Reduce over the 16 columns of this head (lanes within each 16-lane group),
    // then store rows (C layout: VGPR r -> M = r + 8*(lane>>4)).
#pragma unroll
    for (int r = 0; r < 8; ++r) {
        pre0[r] = red16(pre0[r]);
        pre1[r] = red16(pre1[r]);
        pre2[r] = red16(pre2[r]);
        pre3[r] = red16(pre3[r]);
    }
    if (lrow == 0) {
#pragma unroll
        for (int r = 0; r < 8; ++r) {
            long m = r + 8 * lhi;
            long ea = e0 + m;
            long eb = e0 + 16 + m;
            long ec = e0 + 32 + m;
            long ed = e0 + 48 + m;
            if (ea < E) pre[ea * HEADS + head] = 0.25f * pre0[r]; // * K_C^-0.5
            if (eb < E) pre[eb * HEADS + head] = 0.25f * pre1[r];
            if (ec < E) pre[ec * HEADS + head] = 0.25f * pre2[r];
            if (ed < E) pre[ed * HEADS + head] = 0.25f * pre3[r];
        }
    }
}

// ---------------- segment softmax ----------------
__global__ void seg_init(float* nmax, float* nsum, int n) {
    int i = blockIdx.x * blockDim.x + threadIdx.x;
    if (i < n) { nmax[i] = -INFINITY; nsum[i] = 0.f; }
}
__global__ void seg_max(const float* __restrict__ pre, const int* __restrict__ index,
                        float* __restrict__ nmax, int eh) {
    int i = blockIdx.x * blockDim.x + threadIdx.x;
    if (i < eh) {
        int e = i >> 3, h = i & 7;
        atomicMaxF(&nmax[index[e] * HEADS + h], pre[i]);
    }
}
__global__ void seg_expsum(const float* __restrict__ pre, const int* __restrict__ index,
                           const float* __restrict__ nmax, float* __restrict__ nsum,
                           float* __restrict__ out, int eh) {
    int i = blockIdx.x * blockDim.x + threadIdx.x;
    if (i < eh) {
        int e = i >> 3, h = i & 7;
        float ex = expf(pre[i] - nmax[index[e] * HEADS + h]);
        out[i] = ex;
        atomicAdd(&nsum[index[e] * HEADS + h], ex);
    }
}
__global__ void seg_norm(const int* __restrict__ index, const float* __restrict__ nsum,
                         float* __restrict__ out, int eh) {
    int i = blockIdx.x * blockDim.x + threadIdx.x;
    if (i < eh) {
        int e = i >> 3, h = i & 7;
        out[i] = out[i] / (nsum[index[e] * HEADS + h] + 1e-16f);
    }
}

// ---------------- launcher ----------------
extern "C" void kernel_launch(void* const* d_in, const int* in_sizes, int n_in,
                              void* d_out, int out_size, void* d_ws, size_t ws_size,
                              hipStream_t stream) {
    const float* xq    = (const float*)d_in[0];
    const float* xk    = (const float*)d_in[1];
    const float* Wq    = (const float*)d_in[2];
    const float* Wk    = (const float*)d_in[3];
    const int*   index = (const int*)d_in[4];
    const int E = in_sizes[0] / (NORD * IN_C);   // 100000

    // workspace carve-up
    unsigned short* wpack = (unsigned short*)d_ws;               // 950,272 B
    size_t offPre = (size_t)1 << 20;
    float* pre = (float*)((char*)d_ws + offPre);                 // E*8*4 B
    size_t offMax = offPre + (size_t)E * HEADS * sizeof(float);
    offMax = (offMax + 255) & ~(size_t)255;
    float* nmax = (float*)((char*)d_ws + offMax);                // NUM_NODES*8*4 B
    size_t offSum = offMax + (size_t)NUM_NODES * HEADS * sizeof(float);
    float* nsum = (float*)((char*)d_ws + offSum);
    float* out  = (float*)d_out;

    {
        int total = NPROJ * NENT * OUT_C * IN_C;
        so2_pack_w<<<(total + 255) / 256, 256, 0, stream>>>(Wq, Wk, wpack);
    }
    {
        int total = NUM_NODES * HEADS;
        seg_init<<<(total + 255) / 256, 256, 0, stream>>>(nmax, nsum, total);
    }
    {
        int tiles = (E + TILE_M - 1) / TILE_M; // E = 100000 -> 1563 (tail guarded)
        so2_attn_pre<<<tiles, 256, 0, stream>>>(xq, xk, wpack, pre, E);
    }
    int eh = E * HEADS;
    seg_max<<<(eh + 255) / 256, 256, 0, stream>>>(pre, index, nmax, eh);
    seg_expsum<<<(eh + 255) / 256, 256, 0, stream>>>(pre, index, nmax, nsum, out, eh);
    seg_norm<<<(eh + 255) / 256, 256, 0, stream>>>(index, nsum, out, eh);
}
